// GemLiteLinearTriton_46265387712855
// MI455X (gfx1250) — matrix operand order repaired and checked by reading, not compiled, not run on hardware
//
#include <hip/hip_runtime.h>
#include <stdint.h>

// A16W4 grouped asymmetric dequant GEMM for MI455X (gfx1250, wave32, WMMA).
// out[M,O] = x[M,K] @ ((W_q - zeros) * scales)[O,K]^T + bias
// M=256, K=4096, O=14336, group=64.
//
// Roofline: 30.1 GFLOP vs 235 MB of int32 weight codes -> memory bound,
// ~10us floor at 23.3 TB/s. BM=256 (all tokens) so W_q is streamed from HBM
// exactly once; x (4 MB) lives in the 192 MB L2. x is pre-converted to f16
// once (d_ws) and DMA'd into LDS with global_load_async_to_lds_b128.

typedef __attribute__((ext_vector_type(16))) _Float16 v16h;
typedef __attribute__((ext_vector_type(8)))  _Float16 v8h;
typedef __attribute__((ext_vector_type(4)))  _Float16 v4h;
typedef __attribute__((ext_vector_type(8)))  float    v8f;
typedef __attribute__((ext_vector_type(4)))  float    v4f;
typedef __attribute__((ext_vector_type(4)))  int      v4i;

#define M_TOK 256
#define K_DIM 4096
#define O_DIM 14336
#define GRP   64
#define NGRP  (K_DIM / GRP)   // 64 groups

#define BM 256                // all tokens in one block -> W_q streamed once from HBM
#define BN 64                 // output-feature tile
#define BK 32                 // k-step (never crosses a scale group)
#define NT (K_DIM / BK)       // 128 k-steps
#define LDA 40                // padded LDS row stride in halves (80B: conflict-free, 16B aligned)
#define LDB 40
#define THREADS 256           // 8 wave32s

static __device__ __forceinline__ void wait_async0() {
#if __has_builtin(__builtin_amdgcn_s_wait_asynccnt)
    __builtin_amdgcn_s_wait_asynccnt(0);
#else
    asm volatile("s_wait_asynccnt 0x0" ::: "memory");
#endif
}

// Pre-pass: x f32 -> f16 into workspace (done once; the GEMM reads it 224x).
__global__ __launch_bounds__(THREADS)
void x_to_f16(const float* __restrict__ x, _Float16* __restrict__ xh) {
    const int i = (blockIdx.x * THREADS + threadIdx.x) * 4;
    v4f v = *(const v4f*)(x + i);
    v4h h;
#pragma unroll
    for (int j = 0; j < 4; ++j) h[j] = (_Float16)v[j];
    *(v4h*)(xh + i) = h;
}

template <bool XF16>
__global__ __launch_bounds__(THREADS, 1)
void a16w4_dequant_gemm_wmma(const void*  __restrict__ xin,
                             const int*   __restrict__ wq,
                             const float* __restrict__ scales,
                             const float* __restrict__ zeros,
                             const float* __restrict__ bias,
                             float*       __restrict__ out)
{
    __shared__ _Float16 As[2][BM * LDA];   // x tile, [m][k] f16, double buffered
    __shared__ _Float16 Bs[2][BN * LDB];   // dequantized W tile, [n][k] f16

    const int t    = threadIdx.x;
    const int lane = t & 31;
    const int wave = t >> 5;
    const int l16  = lane & 15;
    const int lh   = lane >> 4;        // half-lane select (wave32 WMMA layout)
    const int wm   = (wave & 3) * 64;  // 4 waves tile M (4 x 64 = 256)
    const int wn   = (wave >> 2) * 32; // 2 waves tile N (2 x 32 = 64)

    const int nblk = blockIdx.x * BN;

    // Staging map: thread t owns x row t (32 halves / 32 floats per k-step);
    // W rows: thread t loads row (t&63), 8 int32 codes at col ((t>>6)*8).
    const int wrow = t & 63;
    const int wcol = (t >> 6) * 8;

    v8f acc[4][2];
#pragma unroll
    for (int i = 0; i < 4; ++i)
#pragma unroll
        for (int j = 0; j < 2; ++j)
#pragma unroll
            for (int r = 0; r < 8; ++r)
                acc[i][j][r] = 0.0f;

    v4f  xr[8];          // only used by the XF16=false fallback
    v4i  wr[2];
    float sV, zV;

    // --- x tile staging, f16 fast path: async DMA straight into LDS ---
    auto xasync = [&](int kt, int buf) {
        const _Float16* gp = (const _Float16*)xin + (size_t)t * K_DIM + kt * BK;
        const unsigned  lds = (unsigned)(uintptr_t)&As[buf][t * LDA];
        const uint64_t  ga  = (uint64_t)(uintptr_t)gp;
        asm volatile(
            "global_load_async_to_lds_b128 %0, %1, off\n\t"
            "global_load_async_to_lds_b128 %0, %1, off offset:16\n\t"
            "global_load_async_to_lds_b128 %0, %1, off offset:32\n\t"
            "global_load_async_to_lds_b128 %0, %1, off offset:48"
            :: "v"(lds), "v"(ga) : "memory");
    };

    // --- global loads for k-tile kt into registers (W always; x only in fallback) ---
    auto gload = [&](int kt) {
        const int k0 = kt * BK;
        if constexpr (!XF16) {
            const v4f* xp = (const v4f*)((const float*)xin + (size_t)t * K_DIM + k0);
#pragma unroll
            for (int i = 0; i < 8; ++i) xr[i] = xp[i];
        }
        const v4i* wp = (const v4i*)(wq + (size_t)(nblk + wrow) * K_DIM + k0 + wcol);
        wr[0] = wp[0];
        wr[1] = wp[1];
        const int g = k0 / GRP;                 // group index for this k-tile
        sV = scales[(size_t)(nblk + wrow) * NGRP + g];
        zV = zeros [(size_t)(nblk + wrow) * NGRP + g];
    };

    // --- dequantize/convert registers into LDS buffer `buf` ---
    auto stage = [&](int buf) {
        if constexpr (!XF16) {
            v8h* ap = (v8h*)&As[buf][t * LDA];
#pragma unroll
            for (int i = 0; i < 4; ++i) {
                v8h h;
#pragma unroll
                for (int j = 0; j < 4; ++j) {
                    h[j]     = (_Float16)xr[2 * i][j];
                    h[4 + j] = (_Float16)xr[2 * i + 1][j];
                }
                ap[i] = h;
            }
        }
        v8h wh;
#pragma unroll
        for (int j = 0; j < 4; ++j) {
            wh[j]     = (_Float16)(((float)wr[0][j] - zV) * sV);
            wh[4 + j] = (_Float16)(((float)wr[1][j] - zV) * sV);
        }
        *(v8h*)&Bs[buf][wrow * LDB + wcol] = wh; // ds_store_b128
    };

    // --- one K=32 WMMA step over buffer `buf` ---
    auto compute = [&](int buf) {
        // B fragments: 32x16 f16; lanes 0-15 hold K=0..15, lanes 16-31 hold K=16..31.
        v16h bfr[2];
#pragma unroll
        for (int fn = 0; fn < 2; ++fn) {
            const v8h* p = (const v8h*)&Bs[buf][(wn + fn * 16 + l16) * LDB + lh * 16];
            v8h lo = p[0], hi = p[1];
#pragma unroll
            for (int i = 0; i < 8; ++i) { bfr[fn][i] = lo[i]; bfr[fn][8 + i] = hi[i]; }
        }
        // A fragments: 16x32 f16; lanes 0-15 hold K=0..15, lanes 16-31 hold K=16..31.
#pragma unroll
        for (int fm = 0; fm < 4; ++fm) {
            const _Float16* p = &As[buf][(wm + fm * 16 + l16) * LDA + lh * 16];
            v8h lo = *(const v8h*)p;
            v8h hi = *(const v8h*)(p + 8);
            v16h a;
#pragma unroll
            for (int i = 0; i < 8; ++i) { a[i] = lo[i]; a[8 + i] = hi[i]; }
#pragma unroll
            for (int fn = 0; fn < 2; ++fn)
                acc[fm][fn] = __builtin_amdgcn_wmma_f32_16x16x32_f16(
                    /*neg_a=*/false, a, /*neg_b=*/false, bfr[fn],
                    /*c_mod=*/(short)0, acc[fm][fn],
                    /*reuse_a=*/false, /*reuse_b=*/false);
        }
    };

    // ---- software-pipelined main loop (double-buffered LDS) ----
    if constexpr (XF16) xasync(0, 0);
    gload(0);
    stage(0);
    if constexpr (XF16) wait_async0();
    __syncthreads();
#pragma unroll 2
    for (int kt = 0; kt < NT; ++kt) {
        if (kt + 1 < NT) {
            if constexpr (XF16) xasync(kt + 1, (kt + 1) & 1); // DMA next x tile early
            gload(kt + 1);                                    // prefetch next W tile into regs
        }
        compute(kt & 1);                                      // WMMA on current buffer
        if (kt + 1 < NT) stage((kt + 1) & 1);                 // dequant W into other buffer
        if constexpr (XF16) wait_async0();                    // async x writes landed
        __syncthreads();
    }

    // ---- epilogue: D layout row = VGPR idx + 8*(lane>=16), col = lane%16 ----
#pragma unroll
    for (int fn = 0; fn < 2; ++fn) {
        const int col = nblk + wn + fn * 16 + l16;
        const float bv = bias[col];
#pragma unroll
        for (int fm = 0; fm < 4; ++fm) {
            const int row0 = wm + fm * 16 + lh * 8;
            float* op = out + (size_t)row0 * O_DIM + col;
#pragma unroll
            for (int r = 0; r < 8; ++r)
                op[(size_t)r * O_DIM] = acc[fm][fn][r] + bv;
        }
    }
}

extern "C" void kernel_launch(void* const* d_in, const int* in_sizes, int n_in,
                              void* d_out, int out_size, void* d_ws, size_t ws_size,
                              hipStream_t stream) {
    const float* x      = (const float*)d_in[0];
    const int*   wq     = (const int*)  d_in[1];
    const float* scales = (const float*)d_in[2];
    const float* zeros  = (const float*)d_in[3];
    const float* bias   = (const float*)d_in[4];
    float*       out    = (float*)d_out;

    dim3 grid(O_DIM / BN, 1, 1);   // 224 workgroups
    dim3 block(THREADS, 1, 1);

    const size_t xh_bytes = (size_t)M_TOK * K_DIM * sizeof(_Float16); // 2 MB
    if (ws_size >= xh_bytes) {
        _Float16* xh = (_Float16*)d_ws;
        x_to_f16<<<dim3((M_TOK * K_DIM) / (THREADS * 4)), block, 0, stream>>>(x, xh);
        a16w4_dequant_gemm_wmma<true><<<grid, block, 0, stream>>>(
            (const void*)xh, wq, scales, zeros, bias, out);
    } else {
        a16w4_dequant_gemm_wmma<false><<<grid, block, 0, stream>>>(
            (const void*)x, wq, scales, zeros, bias, out);
    }
}
